// EncNP_41188736369112
// MI455X (gfx1250) — compile-verified, matrix operand order checked by reading, not run on hardware
//
#include <hip/hip_runtime.h>
#include <hip/hip_bf16.h>
#include <math.h>

typedef __attribute__((ext_vector_type(16))) _Float16 v16h;
typedef __attribute__((ext_vector_type(8)))  float    v8f;

#define ALPHA_C 1000.0f
#define BETA_C  100.0f
#define KNEI    24

#if __has_builtin(__builtin_amdgcn_global_load_async_to_lds_b32) && \
    __has_builtin(__builtin_amdgcn_s_wait_asynccnt)
#define HAVE_ASYNC_LDS 1
#else
#define HAVE_ASYNC_LDS 0
#endif

typedef __attribute__((address_space(1))) int g_int;
typedef __attribute__((address_space(3))) int l_int;

// ---------------------------------------------------------------------------
// 1) tr_pos: x0[b, c, n], c = i*28 + 2f + s  (feat=14), in_dim=3, out=84
// ---------------------------------------------------------------------------
__global__ void tr_pos_kernel(const float* __restrict__ xyz, float* __restrict__ x0,
                              int B, int Np) {
  int total = B * 84 * Np;
  int t = blockIdx.x * blockDim.x + threadIdx.x;
  if (t >= total) return;
  int n = t % Np;
  int c = (t / Np) % 84;
  int b = t / (84 * Np);
  int i = c / 28;
  int rem = c % 28;
  int f = rem >> 1;
  int s = rem & 1;
  float tc = xyz[((size_t)b * Np + n) * 3 + i];
  float scale = BETA_C * powf(ALPHA_C, -(float)f / 14.0f);
  float v = tc * scale;
  x0[t] = s ? cosf(v) : sinf(v);
}

// ---------------------------------------------------------------------------
// 2) FPS: one block per batch, dist array in LDS (dynamic)
// ---------------------------------------------------------------------------
__global__ void fps_kernel(const float* __restrict__ xyz, int* __restrict__ out_idx,
                           int Np, int G) {
  extern __shared__ unsigned char fps_smem[];
  float* dist = (float*)fps_smem;           // Np
  float* rv   = dist + Np;                  // 256
  int*   ri   = (int*)(rv + 256);           // 256
  int b = blockIdx.x;
  int tid = threadIdx.x;
  const float* p = xyz + (size_t)b * Np * 3;
  for (int n = tid; n < Np; n += 256) dist[n] = 1e10f;
  __syncthreads();
  int far = 0;
  for (int it = 0; it < G; ++it) {
    if (tid == 0) out_idx[(size_t)b * G + it] = far;
    float cx = p[far * 3 + 0], cy = p[far * 3 + 1], cz = p[far * 3 + 2];
    float lv = -1.0f; int li = 0;
    for (int n = tid; n < Np; n += 256) {
      float dx = p[n * 3 + 0] - cx, dy = p[n * 3 + 1] - cy, dz = p[n * 3 + 2] - cz;
      float d = dx * dx + dy * dy + dz * dz;
      float nd = fminf(dist[n], d);
      dist[n] = nd;
      if (nd > lv) { lv = nd; li = n; }
    }
    rv[tid] = lv; ri[tid] = li;
    __syncthreads();
    for (int st = 128; st > 0; st >>= 1) {
      if (tid < st) {
        if (rv[tid + st] > rv[tid] ||
            (rv[tid + st] == rv[tid] && ri[tid + st] < ri[tid])) {
          rv[tid] = rv[tid + st]; ri[tid] = ri[tid + st];
        }
      }
      __syncthreads();
    }
    int nf = ri[0];
    __syncthreads();
    far = nf;
  }
}

// ---------------------------------------------------------------------------
// 3) gather centers: lc_xyz (B,G,3), lc_x (B,G,Cin)
// ---------------------------------------------------------------------------
__global__ void gather_centers_kernel(const float* __restrict__ xyz,
                                      const float* __restrict__ x_in,
                                      const int* __restrict__ fps_idx,
                                      float* __restrict__ lc_xyz,
                                      float* __restrict__ lc_x,
                                      int B, int G, int Np, int Cin) {
  int t = blockIdx.x * blockDim.x + threadIdx.x;
  if (t >= B * G) return;
  int b = t / G;
  int idx = fps_idx[t];
  for (int j = 0; j < 3; ++j)
    lc_xyz[(size_t)t * 3 + j] = xyz[((size_t)b * Np + idx) * 3 + j];
  for (int c = 0; c < Cin; ++c)
    lc_x[(size_t)t * Cin + c] = x_in[((size_t)b * Cin + c) * Np + idx];
}

// ---------------------------------------------------------------------------
// 4) Gram matrix S[g,n] = lc_xyz[g] . xyz[n] via WMMA f32_16x16x32_f16.
//    One wave per 16-row g tile; K padded 3 -> 32 with zeros.
//    n-tile loop unrolled 4x: independent WMMAs in flight (no D->A/B RAW
//    hazard stalls; compiler schedules 4 v_wmma back-to-back).
// ---------------------------------------------------------------------------
__global__ void gram_kernel(const float* __restrict__ lc_xyz,  // (G,3) batch base
                            const float* __restrict__ xyz,     // (Np,3) batch base
                            float* __restrict__ S,             // (G,Np)
                            int G, int Np) {
  int gt = blockIdx.x;          // g tile
  int lane = threadIdx.x;       // 0..31 (one wave)
  int row = lane & 15;
  int g = gt * 16 + row;
  v16h a = {};
  if (lane < 16) {
    a[0] = (_Float16)lc_xyz[(size_t)g * 3 + 0];
    a[1] = (_Float16)lc_xyz[(size_t)g * 3 + 1];
    a[2] = (_Float16)lc_xyz[(size_t)g * 3 + 2];
  }
  int ntiles = Np / 16;
#pragma unroll 4
  for (int nt = 0; nt < ntiles; ++nt) {
    int n = nt * 16 + row;
    v16h bf = {};
    if (lane < 16) {
      bf[0] = (_Float16)xyz[(size_t)n * 3 + 0];
      bf[1] = (_Float16)xyz[(size_t)n * 3 + 1];
      bf[2] = (_Float16)xyz[(size_t)n * 3 + 2];
    }
    v8f c = {};
    c = __builtin_amdgcn_wmma_f32_16x16x32_f16(false, a, false, bf,
                                               (short)0, c, false, false);
    // D layout: lane = n-col (lane&15), rows m = r + 8*(lane>=16)
    int ncol = nt * 16 + (lane & 15);
    int mbase = (lane < 16) ? 0 : 8;
#pragma unroll
    for (int r = 0; r < 8; ++r) {
      int gm = gt * 16 + mbase + r;
      S[(size_t)gm * Np + ncol] = c[r];
    }
  }
}

// ---------------------------------------------------------------------------
// 5) top-k (k=24). One wave per g. The S row is staged into LDS ONCE via the
//    gfx1250 async-to-LDS path (fallback: plain staging), transformed in
//    place to d(n) = |xyz[n]|^2 - 2 S[g,n], then 24 wave-min selection passes
//    run entirely out of LDS (vs 24x global re-reads).
// ---------------------------------------------------------------------------
__device__ inline unsigned long long wave_min_u64(unsigned long long v) {
#pragma unroll
  for (int off = 16; off > 0; off >>= 1) {
    unsigned lo = (unsigned)v, hi = (unsigned)(v >> 32);
    unsigned olo = (unsigned)__shfl_xor((int)lo, off, 32);
    unsigned ohi = (unsigned)__shfl_xor((int)hi, off, 32);
    unsigned long long o = ((unsigned long long)ohi << 32) | olo;
    if (o < v) v = o;
  }
  return v;
}

__global__ void topk_kernel(const float* __restrict__ xyz,      // batch base (Np,3)
                            const float* __restrict__ S,        // (G,Np)
                            int* __restrict__ knn_idx,          // batch base (G,K)
                            float* __restrict__ knn_xyz,        // batch base (G,K,3)
                            int Np, int G) {
  extern __shared__ float lds_d[];          // Np floats
  int g = blockIdx.x;
  int lane = threadIdx.x;
  const float* Srow = S + (size_t)g * Np;

  // Stage S row into LDS.
#if HAVE_ASYNC_LDS
  for (int n = lane; n < Np; n += 32) {
    __builtin_amdgcn_global_load_async_to_lds_b32(
        (g_int*)(Srow + n),
        (l_int*)(lds_d + n),
        0, 0);
  }
  __builtin_amdgcn_s_wait_asynccnt(0);
#else
  for (int n = lane; n < Np; n += 32) {
    __builtin_prefetch(&Srow[n + 256], 0, 0);
    lds_d[n] = Srow[n];
  }
#endif
  __syncthreads();

  // In-place transform: d(n) = |xyz[n]|^2 - 2*S[g,n]
  for (int n = lane; n < Np; n += 32) {
    float x0 = xyz[(size_t)n * 3 + 0];
    float x1 = xyz[(size_t)n * 3 + 1];
    float x2 = xyz[(size_t)n * 3 + 2];
    lds_d[n] = x0 * x0 + x1 * x1 + x2 * x2 - 2.0f * lds_d[n];
  }
  __syncthreads();

  unsigned long long prev = 0ull;
  for (int kk = 0; kk < KNEI; ++kk) {
    unsigned long long best = ~0ull;
    for (int n = lane; n < Np; n += 32) {
      float d = lds_d[n];
      unsigned u = __float_as_uint(d);
      u = (u & 0x80000000u) ? ~u : (u | 0x80000000u);
      unsigned long long key = ((unsigned long long)u << 32) | (unsigned)n;
      if (key >= prev && key < best) best = key;
    }
    best = wave_min_u64(best);
    if (lane == 0) {
      int n = (int)(best & 0xffffffffull);
      knn_idx[(size_t)g * KNEI + kk] = n;
      knn_xyz[((size_t)g * KNEI + kk) * 3 + 0] = xyz[(size_t)n * 3 + 0];
      knn_xyz[((size_t)g * KNEI + kk) * 3 + 1] = xyz[(size_t)n * 3 + 1];
      knn_xyz[((size_t)g * KNEI + kk) * 3 + 2] = xyz[(size_t)n * 3 + 2];
    }
    prev = best + 1ull;
  }
}

// ---------------------------------------------------------------------------
// 6) global std reductions (normalize() uses a GLOBAL scalar std, ddof=1)
//    acc[0,1] = sum,sumsq of (knn_x - lc_x); acc[2,3] same for xyz;
//    acc[4,5] = resulting stds
// ---------------------------------------------------------------------------
__global__ void init_acc_kernel(float* acc) {
  if (threadIdx.x < 8) acc[threadIdx.x] = 0.0f;
}

__global__ void reduce_diff_x_kernel(const float* __restrict__ x_in,
                                     const float* __restrict__ lc_x,
                                     const int* __restrict__ knn_idx, // (B,G,K)
                                     float* __restrict__ acc,
                                     int B, int G, int Np, int Cin) {
  __shared__ float ss[256], sq[256];
  size_t total = (size_t)B * G * KNEI * Cin;
  float s = 0.0f, q = 0.0f;
  for (size_t e = (size_t)blockIdx.x * blockDim.x + threadIdx.x; e < total;
       e += (size_t)gridDim.x * blockDim.x) {
    int c = (int)(e % Cin);
    size_t r = e / Cin;
    int k = (int)(r % KNEI);
    size_t r2 = r / KNEI;
    int g = (int)(r2 % G);
    int b = (int)(r2 / G);
    int idx = knn_idx[((size_t)b * G + g) * KNEI + k];
    float v = x_in[((size_t)b * Cin + c) * Np + idx] -
              lc_x[((size_t)b * G + g) * Cin + c];
    s += v; q += v * v;
  }
  int tid = threadIdx.x;
  ss[tid] = s; sq[tid] = q;
  __syncthreads();
  for (int st = 128; st > 0; st >>= 1) {
    if (tid < st) { ss[tid] += ss[tid + st]; sq[tid] += sq[tid + st]; }
    __syncthreads();
  }
  if (tid == 0) { atomicAdd(&acc[0], ss[0]); atomicAdd(&acc[1], sq[0]); }
}

__global__ void reduce_diff_xyz_kernel(const float* __restrict__ knn_xyz, // (B,G,K,3)
                                       const float* __restrict__ lc_xyz,  // (B,G,3)
                                       float* __restrict__ acc, int B, int G) {
  __shared__ float ss[256], sq[256];
  size_t total = (size_t)B * G * KNEI * 3;
  float s = 0.0f, q = 0.0f;
  for (size_t e = (size_t)blockIdx.x * blockDim.x + threadIdx.x; e < total;
       e += (size_t)gridDim.x * blockDim.x) {
    int j = (int)(e % 3);
    size_t r = e / 3;                 // (b*G+g)*K + k
    size_t bg = r / KNEI;
    float v = knn_xyz[e] - lc_xyz[bg * 3 + j];
    s += v; q += v * v;
  }
  int tid = threadIdx.x;
  ss[tid] = s; sq[tid] = q;
  __syncthreads();
  for (int st = 128; st > 0; st >>= 1) {
    if (tid < st) { ss[tid] += ss[tid + st]; sq[tid] += sq[tid + st]; }
    __syncthreads();
  }
  if (tid == 0) { atomicAdd(&acc[2], ss[0]); atomicAdd(&acc[3], sq[0]); }
}

__global__ void finalize_std_kernel(float* acc, float cntx, float cntz) {
  if (threadIdx.x == 0) {
    float vx = (acc[1] - acc[0] * acc[0] / cntx) / (cntx - 1.0f);
    float vz = (acc[3] - acc[2] * acc[2] / cntz) / (cntz - 1.0f);
    acc[4] = sqrtf(fmaxf(vx, 0.0f));
    acc[5] = sqrtf(fmaxf(vz, 0.0f));
  }
}

// ---------------------------------------------------------------------------
// 7) LGA + pooling. k_anp(w)+w.mean(-1) == 2*mean_k(w) (e is const along k).
//    out[b,d,g] = 2/24 * sum_k (feat(d,k) + pos(d,k)) * pos(d,k)
// ---------------------------------------------------------------------------
__global__ void lga_kernel(const float* __restrict__ x_in,    // (B,Cin,Np)
                           const float* __restrict__ lc_xyz,  // (B,G,3)
                           const float* __restrict__ lc_x,    // (B,G,Cin)
                           const float* __restrict__ knn_xyz, // (B,G,K,3)
                           const int* __restrict__ knn_idx,   // (B,G,K)
                           const float* __restrict__ acc,     // stds at [4],[5]
                           float* __restrict__ out,           // (B,Cout,G)
                           int B, int G, int Np, int Cin, int Cout) {
  __shared__ float s_k[3][KNEI];    // normalized knn xyz
  __shared__ float s_cr[3][KNEI];   // cross(kxyz_n, lc)
  __shared__ float s_dt[KNEI];      // dot(kxyz_n, lc)
  __shared__ int   s_idx[KNEI];
  __shared__ float s_lc[3];
  int bg = blockIdx.x;
  int b = bg / G, g = bg % G;
  int tid = threadIdx.x;
  float stdx = acc[4] + 1e-5f;
  float stdz = acc[5] + 1e-5f;
  if (tid < 3) s_lc[tid] = lc_xyz[(size_t)bg * 3 + tid];
  if (tid < KNEI) s_idx[tid] = knn_idx[(size_t)bg * KNEI + tid];
  __syncthreads();
  if (tid < KNEI) {
    float k0 = (knn_xyz[((size_t)bg * KNEI + tid) * 3 + 0] - s_lc[0]) / stdz;
    float k1 = (knn_xyz[((size_t)bg * KNEI + tid) * 3 + 1] - s_lc[1]) / stdz;
    float k2 = (knn_xyz[((size_t)bg * KNEI + tid) * 3 + 2] - s_lc[2]) / stdz;
    s_k[0][tid] = k0; s_k[1][tid] = k1; s_k[2][tid] = k2;
    s_cr[0][tid] = k1 * s_lc[2] - k2 * s_lc[1];
    s_cr[1][tid] = k2 * s_lc[0] - k0 * s_lc[2];
    s_cr[2][tid] = k0 * s_lc[1] - k1 * s_lc[0];
    s_dt[tid] = k0 * s_lc[0] + k1 * s_lc[1] + k2 * s_lc[2];
  }
  __syncthreads();
  int feat = Cout / 6;
  int twoFeat = 2 * feat;
  for (int d = tid; d < Cout; d += blockDim.x) {
    int i = d / twoFeat;
    int rem = d - i * twoFeat;
    int f = rem >> 1;
    int s = rem & 1;
    float scale = BETA_C * powf(ALPHA_C, -(float)f / (float)feat);
    float argc = s_lc[i] * scale;
    float Ec = s ? cosf(argc) : sinf(argc);
    float lcxd = (d < Cin) ? lc_x[(size_t)bg * Cin + d] : 0.0f;
    float accum = 0.0f;
    for (int k = 0; k < KNEI; ++k) {
      float arg = s_k[i][k] * scale;
      float Ek = s ? cosf(arg) : sinf(arg);
      float pos = Ek + Ec;
      float fv;
      if (d < Cin) {
        fv = (x_in[((size_t)b * Cin + d) * Np + s_idx[k]] - lcxd) / stdx;
      } else {
        int j = (d - Cin) % 7;
        fv = (j < 3) ? s_k[j][k] : ((j < 6) ? s_cr[j - 3][k] : s_dt[k]);
      }
      accum += (fv + pos) * pos;
    }
    out[((size_t)b * Cout + d) * G + g] = 2.0f * accum / (float)KNEI;
  }
}

// ---------------------------------------------------------------------------
// 8) BatchNorm (over axes b,g) + exact GELU. One block per channel.
// ---------------------------------------------------------------------------
__global__ void bn_gelu_kernel(const float* __restrict__ in,
                               const float* __restrict__ gamma,
                               const float* __restrict__ beta,
                               float* __restrict__ out, int B, int C, int G) {
  __shared__ float ss[256], sq[256];
  __shared__ float s_mean, s_rstd;
  int c = blockIdx.x;
  int tid = threadIdx.x;
  int total = B * G;
  float s = 0.0f, q = 0.0f;
  for (int e = tid; e < total; e += blockDim.x) {
    int b = e / G, g = e % G;
    float v = in[((size_t)b * C + c) * G + g];
    s += v; q += v * v;
  }
  ss[tid] = s; sq[tid] = q;
  __syncthreads();
  for (int st = 128; st > 0; st >>= 1) {
    if (tid < st) { ss[tid] += ss[tid + st]; sq[tid] += sq[tid + st]; }
    __syncthreads();
  }
  if (tid == 0) {
    float mean = ss[0] / (float)total;
    float var = sq[0] / (float)total - mean * mean;   // ddof=0
    s_mean = mean;
    s_rstd = rsqrtf(fmaxf(var, 0.0f) + 1e-5f);
  }
  __syncthreads();
  float mean = s_mean, rstd = s_rstd;
  float ga = gamma[c], be = beta[c];
  for (int e = tid; e < total; e += blockDim.x) {
    int b = e / G, g = e % G;
    size_t o = ((size_t)b * C + c) * G + g;
    float xn = (in[o] - mean) * rstd;
    float y = xn * ga + be;
    out[o] = 0.5f * y * (1.0f + erff(y * 0.70710678118654752f));
  }
}

// ---------------------------------------------------------------------------
// Host orchestration
// ---------------------------------------------------------------------------
extern "C" void kernel_launch(void* const* d_in, const int* in_sizes, int n_in,
                              void* d_out, int out_size, void* d_ws, size_t ws_size,
                              hipStream_t stream) {
  (void)in_sizes; (void)n_in; (void)out_size; (void)ws_size;
  const float* xyz = (const float*)d_in[0];
  const float* bn_g1 = (const float*)d_in[1];
  const float* bn_b1 = (const float*)d_in[2];
  const float* bn_g2 = (const float*)d_in[3];
  const float* bn_b2 = (const float*)d_in[4];
  float* out = (float*)d_out;

  const int B = 4, N1 = 4096, G1 = 2048, C0 = 84, C1 = 168;
  const int N2 = 2048, G2 = 1024, C2 = 336;

  unsigned char* w = (unsigned char*)d_ws;
  auto carve = [&](size_t bytes) -> void* {
    void* p = (void*)w;
    w += (bytes + 255) & ~(size_t)255;
    return p;
  };
  float* x0      = (float*)carve(sizeof(float) * (size_t)B * C0 * N1);
  int*   fps1    = (int*)  carve(sizeof(int)   * (size_t)B * G1);
  float* lcxyz1  = (float*)carve(sizeof(float) * (size_t)B * G1 * 3);
  float* lcx1    = (float*)carve(sizeof(float) * (size_t)B * G1 * C0);
  int*   knn1    = (int*)  carve(sizeof(int)   * (size_t)B * G1 * KNEI);
  float* knnxyz1 = (float*)carve(sizeof(float) * (size_t)B * G1 * KNEI * 3);
  float* Sbuf    = (float*)carve(sizeof(float) * (size_t)G1 * N1); // per-batch slab
  float* lc1     = (float*)carve(sizeof(float) * (size_t)B * C1 * G1);
  float* x1      = (float*)carve(sizeof(float) * (size_t)B * C1 * G1);
  int*   fps2    = (int*)  carve(sizeof(int)   * (size_t)B * G2);
  float* lcxyz2  = (float*)carve(sizeof(float) * (size_t)B * G2 * 3);
  float* lcx2    = (float*)carve(sizeof(float) * (size_t)B * G2 * C1);
  int*   knn2    = (int*)  carve(sizeof(int)   * (size_t)B * G2 * KNEI);
  float* knnxyz2 = (float*)carve(sizeof(float) * (size_t)B * G2 * KNEI * 3);
  float* lc2     = (float*)carve(sizeof(float) * (size_t)B * C2 * G2);
  float* acc     = (float*)carve(sizeof(float) * 8);

  // ---------------- Stage 1 ----------------
  {
    int tot = B * C0 * N1;
    tr_pos_kernel<<<(tot + 255) / 256, 256, 0, stream>>>(xyz, x0, B, N1);

    size_t sh = (size_t)N1 * 4 + 256 * 4 + 256 * 4;
    fps_kernel<<<B, 256, sh, stream>>>(xyz, fps1, N1, G1);

    gather_centers_kernel<<<(B * G1 + 255) / 256, 256, 0, stream>>>(
        xyz, x0, fps1, lcxyz1, lcx1, B, G1, N1, C0);

    for (int b = 0; b < B; ++b) {
      gram_kernel<<<G1 / 16, 32, 0, stream>>>(
          lcxyz1 + (size_t)b * G1 * 3, xyz + (size_t)b * N1 * 3, Sbuf, G1, N1);
      topk_kernel<<<G1, 32, (size_t)N1 * 4, stream>>>(
          xyz + (size_t)b * N1 * 3, Sbuf,
          knn1 + (size_t)b * G1 * KNEI, knnxyz1 + (size_t)b * G1 * KNEI * 3, N1, G1);
    }

    init_acc_kernel<<<1, 32, 0, stream>>>(acc);
    reduce_diff_x_kernel<<<1024, 256, 0, stream>>>(x0, lcx1, knn1, acc, B, G1, N1, C0);
    reduce_diff_xyz_kernel<<<256, 256, 0, stream>>>(knnxyz1, lcxyz1, acc, B, G1);
    finalize_std_kernel<<<1, 32, 0, stream>>>(
        acc, (float)((size_t)B * G1 * KNEI * C0), (float)((size_t)B * G1 * KNEI * 3));

    lga_kernel<<<B * G1, 128, 0, stream>>>(x0, lcxyz1, lcx1, knnxyz1, knn1, acc,
                                           lc1, B, G1, N1, C0, C1);
    bn_gelu_kernel<<<C1, 256, 0, stream>>>(lc1, bn_g1, bn_b1, x1, B, C1, G1);
  }

  // ---------------- Stage 2 (xyz := lcxyz1, N=2048) ----------------
  {
    size_t sh = (size_t)N2 * 4 + 256 * 4 + 256 * 4;
    fps_kernel<<<B, 256, sh, stream>>>(lcxyz1, fps2, N2, G2);

    gather_centers_kernel<<<(B * G2 + 255) / 256, 256, 0, stream>>>(
        lcxyz1, x1, fps2, lcxyz2, lcx2, B, G2, N2, C1);

    for (int b = 0; b < B; ++b) {
      gram_kernel<<<G2 / 16, 32, 0, stream>>>(
          lcxyz2 + (size_t)b * G2 * 3, lcxyz1 + (size_t)b * N2 * 3, Sbuf, G2, N2);
      topk_kernel<<<G2, 32, (size_t)N2 * 4, stream>>>(
          lcxyz1 + (size_t)b * N2 * 3, Sbuf,
          knn2 + (size_t)b * G2 * KNEI, knnxyz2 + (size_t)b * G2 * KNEI * 3, N2, G2);
    }

    init_acc_kernel<<<1, 32, 0, stream>>>(acc);
    reduce_diff_x_kernel<<<1024, 256, 0, stream>>>(x1, lcx2, knn2, acc, B, G2, N2, C1);
    reduce_diff_xyz_kernel<<<256, 256, 0, stream>>>(knnxyz2, lcxyz2, acc, B, G2);
    finalize_std_kernel<<<1, 32, 0, stream>>>(
        acc, (float)((size_t)B * G2 * KNEI * C1), (float)((size_t)B * G2 * KNEI * 3));

    lga_kernel<<<B * G2, 128, 0, stream>>>(x1, lcxyz2, lcx2, knnxyz2, knn2, acc,
                                           lc2, B, G2, N2, C1, C2);
    bn_gelu_kernel<<<C2, 256, 0, stream>>>(lc2, bn_g2, bn_b2, out, B, C2, G2);
  }
}
